// LSTMModelAdvanced_13623636263686
// MI455X (gfx1250) — matrix operand order repaired and checked by reading, not compiled, not run on hardware
//
#include <hip/hip_runtime.h>
#include <cstdint>
#include <cstddef>

typedef __bf16 bf16;
typedef __bf16 v16bf __attribute__((ext_vector_type(16)));
typedef float  v8f   __attribute__((ext_vector_type(8)));

union BF16x16 {
    v16bf v;
    uint4 q[2];
};

// Fast activations: hardware v_rcp_f32 (1 ulp) instead of IEEE divide chains.
__device__ __forceinline__ float sigmoidf_fast(float x) {
    return __builtin_amdgcn_rcpf(1.f + __expf(-x));
}
__device__ __forceinline__ float tanhf_fast(float x) {
    // tanh(x) = 1 - 2/(exp(2x)+1), clamped for exp overflow safety
    float cx = fminf(fmaxf(x, -15.f), 15.f);
    float e = __expf(2.f * cx);
    return 1.f - 2.f * __builtin_amdgcn_rcpf(e + 1.f);
}

// ---------------------------------------------------------------------------
// Prep: build combined bf16 weight matrix Wc[4H][Kc] = [w_ih (zero-padded to
// KI) | w_hh], row-major, plus summed bias bsum[4H] = b_ih + b_hh.
// ---------------------------------------------------------------------------
__global__ void prep_weights(const float* __restrict__ w_ih,
                             const float* __restrict__ w_hh,
                             const float* __restrict__ b_ih,
                             const float* __restrict__ b_hh,
                             bf16* __restrict__ Wc, float* __restrict__ bsum,
                             int fourH, int I, int KI, int Kc) {
    int idx = blockIdx.x * blockDim.x + threadIdx.x;
    int total = fourH * Kc;
    int H = Kc - KI;
    if (idx < total) {
        int n = idx / Kc;
        int k = idx - n * Kc;
        float v;
        if (k < KI) v = (k < I) ? w_ih[n * I + k] : 0.f;
        else        v = w_hh[n * H + (k - KI)];
        Wc[idx] = (bf16)v;
    } else if (idx < total + fourH) {
        int n = idx - total;
        bsum[n] = b_ih[n] + b_hh[n];
    }
}

// ---------------------------------------------------------------------------
// Fused persistent LSTM layer.
//   g[m,n] = [x_t(m,:) | h_{t-1}(m,:)] @ Wc[n,:]^T + bsum[n]
// Each WG owns ROWS = 16*(128/H) batch rows for the whole sequence.
// 8 waves; wave -> (m-tile, 16-col strip replicated across the 4 gates) so
// i/f/g/o for a given (m,j) live in one lane's 4 accumulators.
// Weights (B fragments) are VGPR-resident across all 512 timesteps; the bias
// lives in persistent v8f vectors used as the C operand of the first K-step.
// ---------------------------------------------------------------------------
template <int H, int I, int KI, bool XF32>
__global__ __launch_bounds__(256, 1)
void lstm_layer(const void* __restrict__ xin_v,
                const bf16* __restrict__ Wc,
                const float* __restrict__ bsum,
                bf16* __restrict__ xout) {
    constexpr int Kc     = KI + H;     // combined K (multiple of 32)
    constexpr int NK     = Kc / 32;    // K-steps of 32 (bf16 WMMA)
    constexpr int MT     = 128 / H;    // M tiles per WG
    constexpr int ROWS   = 16 * MT;    // batch rows per WG
    constexpr int STRIPS = H / 16;     // 16-col strips per gate
    constexpr int SL     = 512;        // sequence length

    __shared__ __align__(16) bf16 a_tile[ROWS][Kc];  // [x_t | h_{t-1}]

    const int tid    = threadIdx.x;
    const int wave   = tid >> 5;
    const int lane   = tid & 31;
    const int lane15 = lane & 15;
    const int hi     = lane >> 4;
    const int hi8    = hi * 8;
    const int ws     = wave % STRIPS;      // strip within gate
    const int mt     = wave / STRIPS;      // m-tile
    const int mt16   = mt * 16;
    const int j      = ws * 16 + lane15;   // hidden index this lane owns
    const int gm0    = blockIdx.x * ROWS;  // global batch row base

    // persistent bias vectors (C operand of first K-step WMMA)
    v8f biasv[4];
#pragma unroll
    for (int g = 0; g < 4; ++g) {
        float b = bsum[g * H + j];
#pragma unroll
        for (int r = 0; r < 8; ++r) biasv[g][r] = b;
    }

    // -------- resident B fragments: Wc rows (gate g, column j), all K-steps.
    // B operand (K x 16), this code's k order: lane holds column n=lane15, 16 contiguous K
    // values starting at 32*kk + hi*16, packed 2/VGPR.
    BF16x16 bfrag[NK][4];
#pragma unroll
    for (int g = 0; g < 4; ++g) {
        const bf16* wrow = Wc + (size_t)(g * H + j) * Kc;
#pragma unroll
        for (int kk = 0; kk < NK; ++kk) {
            const bf16* p = wrow + kk * 32 + hi * 16;
            bfrag[kk][g].q[0] = *(const uint4*)(p);
            bfrag[kk][g].q[1] = *(const uint4*)(p + 8);
        }
    }

    // staging helper: load x_t into a_tile[:, 0:KI]
    auto stage_x = [&](int t) {
        if constexpr (XF32) {
            const float* xin = (const float*)xin_v;
            for (int idx = tid; idx < ROWS * KI; idx += 256) {
                int r = idx / KI, k = idx - r * KI;
                float v = (k < I) ? xin[((size_t)(gm0 + r) * SL + t) * I + k] : 0.f;
                a_tile[r][k] = (bf16)v;
            }
        } else {
            const bf16* xin = (const bf16*)xin_v;
            constexpr int CH = KI / 8;  // 16B chunks per row
            for (int idx = tid; idx < ROWS * CH; idx += 256) {
                int r = idx / CH, kb = (idx - r * CH) * 8;
                *(uint4*)&a_tile[r][kb] =
                    *(const uint4*)&xin[((size_t)(gm0 + r) * SL + t) * I + kb];
            }
        }
    };

    // zero h_{-1}
    for (int idx = tid; idx < ROWS * H; idx += 256) {
        int r = idx / H, k = idx - r * H;
        a_tile[r][KI + k] = (bf16)0.f;
    }
    stage_x(0);

    // cell state, VGPR-resident (C/D layout: lane=col j, VGPR r -> row r+hi8)
    float creg[8];
#pragma unroll
    for (int r = 0; r < 8; ++r) creg[r] = 0.f;

    for (int t = 0; t < SL; ++t) {
        __syncthreads();  // staging (x_t, h_{t-1}) visible

        // -------- A fragments: A operand (16 x K), this code's k order: lane = row m,
        // 16 contiguous k starting at 32kk + hi*16 (matches B).
        BF16x16 afrag[NK];
#pragma unroll
        for (int kk = 0; kk < NK; ++kk) {
            const bf16* base = &a_tile[mt16 + lane15][kk * 32 + hi * 16];
            afrag[kk].q[0] = *(const uint4*)(base);
            afrag[kk].q[1] = *(const uint4*)(base + 8);
        }

        __syncthreads();  // all reads done; staging may be overwritten

        if (t + 1 < SL) stage_x(t + 1);
        if (t + 2 < SL && tid < ROWS) {  // prefetch next-next x row
            if constexpr (XF32)
                __builtin_prefetch((const float*)xin_v +
                                   ((size_t)(gm0 + tid) * SL + t + 2) * I, 0, 1);
            else
                __builtin_prefetch((const bf16*)xin_v +
                                   ((size_t)(gm0 + tid) * SL + t + 2) * I, 0, 1);
        }

        // -------- gate pre-activations: acc = A @ B + bias
        // First K-step consumes the persistent bias vector as C (by value),
        // so no per-step accumulator splat is needed.
        v8f acc[4];
#pragma unroll
        for (int g = 0; g < 4; ++g) {
            acc[g] = __builtin_amdgcn_wmma_f32_16x16x32_bf16(
                false, afrag[0].v, false, bfrag[0][g].v,
                (short)0, biasv[g], false, false);
        }
#pragma unroll
        for (int kk = 1; kk < NK; ++kk) {
#pragma unroll
            for (int g = 0; g < 4; ++g) {
                acc[g] = __builtin_amdgcn_wmma_f32_16x16x32_bf16(
                    false, afrag[kk].v, false, bfrag[kk][g].v,
                    (short)0, acc[g], false, false);
            }
        }

        // -------- elementwise LSTM cell (registers only)
#pragma unroll
        for (int r = 0; r < 8; ++r) {
            float iv = sigmoidf_fast(acc[0][r]);
            float fv = sigmoidf_fast(acc[1][r]);
            float gv = tanhf_fast(acc[2][r]);
            float ov = sigmoidf_fast(acc[3][r]);
            float cn = fv * creg[r] + iv * gv;
            creg[r] = cn;
            float hv = ov * tanhf_fast(cn);
            bf16 hb = (bf16)hv;
            int m = r + hi8;
            a_tile[mt16 + m][KI + j] = hb;  // h_t for next step's A
            xout[((size_t)(gm0 + mt16 + m) * SL + t) * H + j] = hb;
        }
    }
}

// ---------------------------------------------------------------------------
// Final head: out[b] = h_last[b,:] @ lin_w + lin_b
// ---------------------------------------------------------------------------
__global__ void final_linear(const bf16* __restrict__ x4,
                             const float* __restrict__ lin_w,
                             const float* __restrict__ lin_b,
                             float* __restrict__ out) {
    int b = blockIdx.x * blockDim.x + threadIdx.x;
    if (b >= 1024) return;
    const bf16* hrow = x4 + ((size_t)b * 512 + 511) * 64;
    float acc = lin_b[0];
#pragma unroll 8
    for (int k = 0; k < 64; ++k) acc += (float)hrow[k] * lin_w[k];
    out[b] = acc;
}

// ---------------------------------------------------------------------------
extern "C" void kernel_launch(void* const* d_in, const int* in_sizes, int n_in,
                              void* d_out, int out_size, void* d_ws, size_t ws_size,
                              hipStream_t stream) {
    (void)in_sizes; (void)n_in; (void)out_size; (void)ws_size;

    const float* input_seq = (const float*)d_in[0];
    const float* w_ih[4], * w_hh[4], * b_ih[4], * b_hh[4];
    for (int l = 0; l < 4; ++l) {
        w_ih[l] = (const float*)d_in[1 + 4 * l];
        w_hh[l] = (const float*)d_in[2 + 4 * l];
        b_ih[l] = (const float*)d_in[3 + 4 * l];
        b_hh[l] = (const float*)d_in[4 + 4 * l];
    }
    const float* lin_w = (const float*)d_in[17];
    const float* lin_b = (const float*)d_in[18];

    // workspace bump allocator
    char* ws = (char*)d_ws;
    auto alloc = [&](size_t bytes) {
        void* p = (void*)ws;
        ws += (bytes + 255) & ~(size_t)255;
        return p;
    };

    // layer geometry: {fourH, I, KI, Kc}
    // L0: H=128 I=8   KI=32  Kc=160 | L1: H=128 I=128 KI=128 Kc=256
    // L2: H=64  I=128 KI=128 Kc=192 | L3: H=64  I=64  KI=64  Kc=128
    bf16* Wc0 = (bf16*)alloc((size_t)512 * 160 * sizeof(bf16));
    bf16* Wc1 = (bf16*)alloc((size_t)512 * 256 * sizeof(bf16));
    bf16* Wc2 = (bf16*)alloc((size_t)256 * 192 * sizeof(bf16));
    bf16* Wc3 = (bf16*)alloc((size_t)256 * 128 * sizeof(bf16));
    float* bs0 = (float*)alloc(512 * sizeof(float));
    float* bs1 = (float*)alloc(512 * sizeof(float));
    float* bs2 = (float*)alloc(256 * sizeof(float));
    float* bs3 = (float*)alloc(256 * sizeof(float));
    const size_t BS = (size_t)1024 * 512;
    bf16* x1 = (bf16*)alloc(BS * 128 * sizeof(bf16));
    bf16* x2 = (bf16*)alloc(BS * 128 * sizeof(bf16));
    bf16* x3 = (bf16*)alloc(BS * 64 * sizeof(bf16));
    bf16* x4 = (bf16*)alloc(BS * 64 * sizeof(bf16));

    auto pgrid = [](int fourH, int Kc) { return (fourH * Kc + fourH + 255) / 256; };
    prep_weights<<<pgrid(512, 160), 256, 0, stream>>>(w_ih[0], w_hh[0], b_ih[0], b_hh[0], Wc0, bs0, 512, 8,   32,  160);
    prep_weights<<<pgrid(512, 256), 256, 0, stream>>>(w_ih[1], w_hh[1], b_ih[1], b_hh[1], Wc1, bs1, 512, 128, 128, 256);
    prep_weights<<<pgrid(256, 192), 256, 0, stream>>>(w_ih[2], w_hh[2], b_ih[2], b_hh[2], Wc2, bs2, 256, 128, 128, 192);
    prep_weights<<<pgrid(256, 128), 256, 0, stream>>>(w_ih[3], w_hh[3], b_ih[3], b_hh[3], Wc3, bs3, 256, 64,  64,  128);

    lstm_layer<128, 8,   32,  true ><<<64, 256, 0, stream>>>(input_seq, Wc0, bs0, x1);
    lstm_layer<128, 128, 128, false><<<64, 256, 0, stream>>>(x1,        Wc1, bs1, x2);
    lstm_layer<64,  128, 128, false><<<32, 256, 0, stream>>>(x2,        Wc2, bs2, x3);
    lstm_layer<64,  64,  64,  false><<<32, 256, 0, stream>>>(x3,        Wc3, bs3, x4);

    final_linear<<<4, 256, 0, stream>>>(x4, lin_w, lin_b, (float*)d_out);
}
